// PostProcessingModule_11965778887099
// MI455X (gfx1250) — matrix operand order repaired and checked by reading, not compiled, not run on hardware
//
#include <hip/hip_runtime.h>
#include <cstdint>
#include <cstddef>

// Problem constants (from reference)
#define TOKENS       32768      // B*S = 8*4096
#define DMODEL       2048
#define NUM_NODES    32
#define NODE_DIM     64
#define TOPKN        16
#define HID          32
#define KDIM         1024       // TOPK * NODE_DIM
// GEMM tiling
#define BLOCK_N      128                         // per workgroup (8 waves x 16)
#define N_STRIPES    (DMODEL / BLOCK_N)          // 16
#define MT_PER_CHUNK 16                          // 16 m-tiles (256 tokens) per WG
#define NUM_CHUNKS   (TOKENS / (16 * MT_PER_CHUNK)) // 128
// LDS: B stripe (256KB) + double-buffered A tile (2 x 32KB) = 320KB (WGP max)
#define GEMM_LDS_BYTES ((BLOCK_N * KDIM + 2 * 16 * KDIM) * 2)

typedef __attribute__((ext_vector_type(16))) __bf16 v16bf;
typedef __attribute__((ext_vector_type(8)))  float  v8f;

__device__ __forceinline__ unsigned short f2bf(float f) {
  unsigned int u = __float_as_uint(f);
  u += 0x7FFFu + ((u >> 16) & 1u);            // round-to-nearest-even
  return (unsigned short)(u >> 16);
}

__device__ __forceinline__ uint4 pack8_bf16(float4 a, float4 b) {
  uint4 o;
  o.x = (unsigned)f2bf(a.x) | ((unsigned)f2bf(a.y) << 16);
  o.y = (unsigned)f2bf(a.z) | ((unsigned)f2bf(a.w) << 16);
  o.z = (unsigned)f2bf(b.x) | ((unsigned)f2bf(b.y) << 16);
  o.w = (unsigned)f2bf(b.z) | ((unsigned)f2bf(b.w) << 16);
  return o;
}

// ---------------------------------------------------------------------------
// K0: wp [1024][2048] f32  ->  wpT [2048][1024] bf16  (tiled LDS transpose)
// ---------------------------------------------------------------------------
__global__ __launch_bounds__(256)
void wp_transpose_kernel(const float* __restrict__ wp,
                         unsigned short* __restrict__ wpT) {
  __shared__ float tile[32][33];
  const int k0 = blockIdx.y * 32;   // row (p) of wp
  const int d0 = blockIdx.x * 32;   // col (d) of wp
  const int tx = threadIdx.x;       // 0..31
  const int ty = threadIdx.y;       // 0..7
#pragma unroll
  for (int i = 0; i < 32; i += 8)
    tile[ty + i][tx] = wp[(size_t)(k0 + ty + i) * DMODEL + d0 + tx];
  __syncthreads();
#pragma unroll
  for (int i = 0; i < 32; i += 8)
    wpT[(size_t)(d0 + ty + i) * KDIM + k0 + tx] = f2bf(tile[tx][ty + i]);
}

// ---------------------------------------------------------------------------
// K1: per-token score MLP + stable top-16 (one wave32 per token, lane = node)
// Softmax is monotonic and b2/TEMP are uniform -> ordering by raw score.
// Two independent accumulator chains -> dual-issue f32 FMA.
// ---------------------------------------------------------------------------
__global__ __launch_bounds__(256)
void score_topk_kernel(const float* __restrict__ gf,
                       const float* __restrict__ w1,
                       const float* __restrict__ b1,
                       const float* __restrict__ w2,
                       unsigned char* __restrict__ idx_out) {
  const int lane = threadIdx.x & 31;
  const int wave = threadIdx.x >> 5;
  const int tok  = blockIdx.x * 8 + wave;

  // lane = node: load this node's 64 dims
  const float4* node4 = (const float4*)(gf + (size_t)tok * DMODEL + lane * NODE_DIM);
  float x[NODE_DIM];
#pragma unroll
  for (int i = 0; i < 16; ++i) {
    float4 v = node4[i];
    x[4*i+0] = v.x; x[4*i+1] = v.y; x[4*i+2] = v.z; x[4*i+3] = v.w;
  }

  float score = 0.0f;
#pragma unroll 1
  for (int j = 0; j < HID; j += 2) {
    float a0 = b1[j];                          // uniform -> scalar loads
    float a1 = b1[j + 1];
#pragma unroll
    for (int d = 0; d < NODE_DIM; ++d) {
      const float xv = x[d];
      a0 = fmaf(xv, w1[d * HID + j],     a0);
      a1 = fmaf(xv, w1[d * HID + j + 1], a1);
    }
    // exact GELU: 0.5*a*(1+erf(a/sqrt(2)))
    float g0 = 0.5f * a0 * (1.0f + erff(a0 * 0.70710678118654752f));
    float g1 = 0.5f * a1 * (1.0f + erff(a1 * 0.70710678118654752f));
    score = fmaf(g0, w2[j], fmaf(g1, w2[j + 1], score));
  }

  // stable descending rank across the 32 lanes (ties: lower index first)
  int rank = 0;
#pragma unroll
  for (int l = 0; l < 32; ++l) {
    float o = __shfl(score, l, 32);
    if ((o > score) || (o == score && l < lane)) ++rank;
  }
  if (rank < TOPKN)
    idx_out[(size_t)tok * TOPKN + rank] = (unsigned char)lane;
}

// ---------------------------------------------------------------------------
// K2: fused gather + GEMM  out[tok][d] = sum_k pooled_bf16[tok][k]*wpT[d][k] + bp[d]
// WG = (stripe of 128 N, chunk of 256 tokens). B stripe async-loaded to LDS
// once; A tiles (gather + f32->bf16) double-buffered under the WMMA loop.
// ---------------------------------------------------------------------------
extern __shared__ char smem_raw[];

__device__ __forceinline__ void gather_a_tile(
    const float* __restrict__ gf, const unsigned char* __restrict__ topk_idx,
    int tok0, int tokLd, int slot, unsigned short* __restrict__ dstTile) {
  const int tok  = tok0 + tokLd;
  const int nidx = topk_idx[(size_t)tok * TOPKN + slot];
  const float4* src = (const float4*)(gf + (size_t)tok * DMODEL + nidx * NODE_DIM);
  uint4* dstA = (uint4*)(dstTile + tokLd * KDIM + slot * NODE_DIM);
#pragma unroll
  for (int i = 0; i < 8; ++i)
    dstA[i] = pack8_bf16(src[2 * i], src[2 * i + 1]);
}

__global__ __launch_bounds__(256)
void gather_gemm_kernel(const float* __restrict__ gf,
                        const unsigned char* __restrict__ topk_idx,
                        const unsigned short* __restrict__ wpT,
                        const float* __restrict__ bp,
                        float* __restrict__ out) {
  unsigned short* Bs  = (unsigned short*)smem_raw;       // [BLOCK_N][KDIM]
  unsigned short* As0 = Bs + BLOCK_N * KDIM;             // [16][KDIM] buffer 0
  unsigned short* As1 = As0 + 16 * KDIM;                 // [16][KDIM] buffer 1

  const int tid   = threadIdx.x;
  const int wave  = tid >> 5;
  const int lane  = tid & 31;
  const int n0    = blockIdx.x * BLOCK_N;  // stripe (fastest dim -> L2 reuse)
  const int chunk = blockIdx.y;

  const int tokLd = tid >> 4;              // gather mapping: 16 threads/token
  const int slot  = tid & 15;              // rank slot (preserves sorted order)

  // ---- B stripe: 128 consecutive rows of wpT are linear in memory.
  //      CDNA5 async copy global->LDS (no VGPR round-trip, tracked by ASYNCcnt)
  {
    const unsigned short* srcB = wpT + (size_t)n0 * KDIM;
    for (int i = tid; i < BLOCK_N * KDIM / 8; i += 256) {
      unsigned lds_off = (unsigned)(unsigned long long)(uintptr_t)(Bs + i * 8);
      unsigned long long ga = (unsigned long long)(uintptr_t)(srcB + i * 8);
      asm volatile("global_load_async_to_lds_b128 %0, %1, off"
                   :: "v"(lds_off), "v"(ga) : "memory");
    }
  }

  // ---- prologue: gather first A tile while async B copies run
  gather_a_tile(gf, topk_idx, chunk * MT_PER_CHUNK * 16, tokLd, slot, As0);

  asm volatile("s_wait_asynccnt 0x0" ::: "memory");
  __syncthreads();

  // per-thread WMMA-fragment constants
  const int hk     = lane >> 4;                 // lane-half
  const int rowA   = lane & 15;                 // A: M row
  const int col    = n0 + wave * 16 + (lane & 15);  // C: N = lane%16
  const int mOff   = hk * 8;                    // C: M = vgpr + 8*(lane>=16)
  const float bias = bp[col];

  const unsigned short* bBase = Bs + (wave * 16 + (lane & 15)) * KDIM;

  for (int mt = 0; mt < MT_PER_CHUNK; ++mt) {
    const int tok0 = (chunk * MT_PER_CHUNK + mt) * 16;
    unsigned short* cur = (mt & 1) ? As1 : As0;
    unsigned short* nxt = (mt & 1) ? As0 : As1;

    // ---- prefetch next A tile into the other buffer (overlaps WMMA loop)
    if (mt + 1 < MT_PER_CHUNK)
      gather_a_tile(gf, topk_idx, tok0 + 16, tokLd, slot, nxt);

    // ---- 16x16 tile, K = 1024 in 32 WMMA steps
    const unsigned short* aBase = cur + rowA * KDIM;
    v8f acc;
#pragma unroll
    for (int r = 0; r < 8; ++r) acc[r] = 0.0f;

#pragma unroll 4
    for (int kk = 0; kk < KDIM; kk += 32) {
      union { uint4 u[2]; v16bf v; } af, bf;
      // A (16x32, bf16): lane-halves hold K {0-7,16-23} / {8-15,24-31}
      af.u[0] = *(const uint4*)(aBase + kk + hk * 8);
      af.u[1] = *(const uint4*)(aBase + kk + 16 + hk * 8);
      // B (32x16, bf16): lane = column; lane-halves hold K {0-7,16-23} / {8-15,24-31}
      bf.u[0] = *(const uint4*)(bBase + kk + hk * 8);
      bf.u[1] = *(const uint4*)(bBase + kk + 16 + hk * 8);
      acc = __builtin_amdgcn_wmma_f32_16x16x32_bf16(
          false, af.v, false, bf.v, (short)0, acc, false, false);
    }

    // ---- store C tile (+bias): lane l -> col, vgpr r -> row mOff+r
    float* orow = out + (size_t)(tok0 + mOff) * DMODEL + col;
#pragma unroll
    for (int r = 0; r < 8; ++r)
      orow[(size_t)r * DMODEL] = acc[r] + bias;

    __syncthreads();   // next-tile gather visible; buffer safe to overwrite
  }
}

// ---------------------------------------------------------------------------
extern "C" void kernel_launch(void* const* d_in, const int* in_sizes, int n_in,
                              void* d_out, int out_size, void* d_ws, size_t ws_size,
                              hipStream_t stream) {
  const float* gf = (const float*)d_in[0];   // [8,4096,2048]
  const float* w1 = (const float*)d_in[1];   // [64,32]
  const float* b1 = (const float*)d_in[2];   // [32]
  const float* w2 = (const float*)d_in[3];   // [32,1]
  // d_in[4] = b2: uniform shift, ordering-invariant -> unused
  const float* wp = (const float*)d_in[5];   // [1024,2048]
  const float* bp = (const float*)d_in[6];   // [2048]
  float* out = (float*)d_out;                // [8,4096,2048]

  // workspace layout: [0,4MB) wpT bf16 ; [4MB, 4.5MB) top-k indices (u8)
  unsigned short* wpT  = (unsigned short*)d_ws;
  unsigned char*  idxb = (unsigned char*)d_ws + (size_t)DMODEL * KDIM * 2;

  (void)in_sizes; (void)n_in; (void)out_size; (void)ws_size;

  hipFuncSetAttribute(reinterpret_cast<const void*>(&gather_gemm_kernel),
                      hipFuncAttributeMaxDynamicSharedMemorySize, GEMM_LDS_BYTES);

  wp_transpose_kernel<<<dim3(DMODEL / 32, KDIM / 32), dim3(32, 8), 0, stream>>>(wp, wpT);
  score_topk_kernel<<<TOKENS / 8, 256, 0, stream>>>(gf, w1, b1, w2, idxb);
  gather_gemm_kernel<<<dim3(N_STRIPES, NUM_CHUNKS), 256, GEMM_LDS_BYTES, stream>>>(
      gf, idxb, wpT, bp, out);
}
